// TriangleAttention_85804856640040
// MI455X (gfx1250) — matrix operand order repaired and checked by reading, not compiled, not run on hardware
//
#include <hip/hip_runtime.h>

// ---------------------------------------------------------------------------
// Types for CDNA5 WMMA (wave32): V_WMMA_F32_16X16X32_F16
// ---------------------------------------------------------------------------
typedef __attribute__((ext_vector_type(16))) _Float16 v16h;
typedef __attribute__((ext_vector_type(8)))  _Float16 v8h;
typedef __attribute__((ext_vector_type(8)))  float    v8f;
typedef __attribute__((ext_vector_type(4)))  unsigned int v4u;
typedef __attribute__((ext_vector_type(8)))  int      v8i;
typedef __attribute__((ext_vector_type(4)))  int      v4i;

#define WMMA_F16(a, b, c) \
    __builtin_amdgcn_wmma_f32_16x16x32_f16(false, (a), false, (b), (short)0, (c), false, false)

// Problem dims
#define BB 256
#define NN 256
#define CC 128
#define HH 4
#define DD 128
#define OO 128

// ---------------------------------------------------------------------------
// Fragment loaders per CDNA5 ISA §7.12.2 (wave32)
// A (16xK, row-major, K contiguous): lane holds row (lane&15); two contiguous
//   8-halfword chunks at k = kbase + half*8 and kbase + 16 + half*8.
// B (Kx16) loaded from a B^T tile (16 rows x K, K contiguous): lane holds
//   column (lane&15); two contiguous 8-halfword chunks at k = kbase + half*8 and kbase + 16 + half*8.
// ---------------------------------------------------------------------------
__device__ __forceinline__ v16h load_fragA(const _Float16* base, int ld, int kbase) {
    const int lane = threadIdx.x & 31;
    const int row  = lane & 15;
    const int half = lane >> 4;
    const _Float16* p0 = base + row * ld + kbase + half * 8;
    v8h lo = *(const v8h*)(p0);
    v8h hi = *(const v8h*)(p0 + 16);
    v16h r;
#pragma unroll
    for (int i = 0; i < 8; ++i) { r[i] = lo[i]; r[i + 8] = hi[i]; }
    return r;
}

__device__ __forceinline__ v16h load_fragB(const _Float16* base, int ld, int kbase) {
    const int lane = threadIdx.x & 31;
    const int col  = lane & 15;
    const int half = lane >> 4;
    const _Float16* p0 = base + col * ld + kbase + half * 8;
    v8h lo = *(const v8h*)(p0);
    v8h hi = *(const v8h*)(p0 + 16);
    v16h r;
#pragma unroll
    for (int i = 0; i < 8; ++i) { r[i] = lo[i]; r[i + 8] = hi[i]; }
    return r;
}

// ---------------------------------------------------------------------------
// Tensor Data Mover: 2D tile (f16 elements) global -> LDS.
// Builds D# groups per CDNA5 ISA 8.3/8.4 and issues TENSOR_LOAD_TO_LDS via
// the 6-arg builtin (this toolchain's arity: g0, g1, g2, g3, extra, cpol).
// Wave-level op (EXEC ignored), tracked by TENSORcnt.
//   row_elems  = tile width  (elements, contiguous, = tensor_dim0 = tile_dim0)
//   rows       = tile height (= tensor_dim1 = tile_dim1)
//   stride     = row stride in elements (tensor_dim0_stride)
// ---------------------------------------------------------------------------
__device__ __forceinline__ void tdm_load_2d_f16(unsigned lds_off, const void* gptr,
                                                unsigned row_elems, unsigned rows,
                                                unsigned stride) {
    unsigned long long ga = (unsigned long long)(size_t)gptr;
    v4u g0;
    g0[0] = 1u;                                           // count=1, user descriptor
    g0[1] = lds_off;                                      // lds_addr (bytes)
    g0[2] = (unsigned)(ga & 0xFFFFFFFFu);                 // global_addr[31:0]
    g0[3] = (unsigned)((ga >> 32) & 0x01FFFFFFu) | (2u << 30);  // addr[56:32] | type=2
    v8i g1;
    g1[0] = 0x00010000;                                   // data_size=1 (2-byte elems)
    g1[1] = (int)((row_elems & 0xFFFFu) << 16);           // tensor_dim0[15:0]
    g1[2] = (int)((row_elems >> 16) | ((rows & 0xFFFFu) << 16));  // dim0 hi | dim1 lo
    g1[3] = (int)((rows >> 16) | (row_elems << 16));      // dim1 hi | tile_dim0
    g1[4] = (int)rows;                                    // tile_dim1 (tile_dim2=0)
    g1[5] = (int)stride;                                  // tensor_dim0_stride[31:0]
    g1[6] = 0;                                            // stride0 hi | stride1 lo
    g1[7] = 0;                                            // stride1 hi
    v4i gz4 = {};                                         // groups 2/3 unused (2D)
    v8i gz8 = {};
    __builtin_amdgcn_tensor_load_to_lds(g0, g1, gz4, gz4, gz8, 0);
}

// ---------------------------------------------------------------------------
// Kernel 0: weight transpose + f32->f16 convert.
// ---------------------------------------------------------------------------
__global__ void prep_weights(const float* __restrict__ qw, const float* __restrict__ kw,
                             const float* __restrict__ vw, const float* __restrict__ gw,
                             const float* __restrict__ ow,
                             _Float16* __restrict__ wqt, _Float16* __restrict__ wkt,
                             _Float16* __restrict__ wvt, _Float16* __restrict__ wgt,
                             _Float16* __restrict__ w2t) {
    int idx = blockIdx.x * blockDim.x + threadIdx.x;      // 0 .. C*H*D-1 (65536)
    const float qscale = 0.08838834764831845f;            // 128^-0.5
    int c  = idx >> 9;                                    // / (H*D)
    int hd = idx & 511;
    int t  = hd * CC + c;
    wqt[t] = (_Float16)(qw[idx] * qscale);
    wkt[t] = (_Float16)(kw[idx]);
    wvt[t] = (_Float16)(vw[idx]);
    wgt[t] = (_Float16)(gw[idx]);
    int hd2 = idx >> 7;                                   // / O
    int o   = idx & 127;
    w2t[o * (HH * DD) + hd2] = (_Float16)(ow[idx]);
}

// ---------------------------------------------------------------------------
// Kernel 1: fused QKV+gate projection. One wave per 16 rows of (B*N).
//   q,k,gate -> (B,H,N,D) f16 ; v -> (B,H,D,N) f16 (transposed for P*V).
// ---------------------------------------------------------------------------
__global__ __launch_bounds__(32) void proj_kernel(
    const float* __restrict__ qd, const float* __restrict__ md,
    const _Float16* __restrict__ wqt, const _Float16* __restrict__ wkt,
    const _Float16* __restrict__ wvt, const _Float16* __restrict__ wgt,
    const float* __restrict__ gbias,
    _Float16* __restrict__ qb, _Float16* __restrict__ kb,
    _Float16* __restrict__ vtb, _Float16* __restrict__ gateb) {
    __shared__ _Float16 xq[16 * CC];
    __shared__ _Float16 xm[16 * CC];

    const int lane = threadIdx.x;
    const size_t r0 = (size_t)blockIdx.x * 16;            // row block in B*N

    const float* qsrc = qd + r0 * CC;
    const float* msrc = md + r0 * CC;
#pragma unroll 4
    for (int i = lane; i < 16 * CC; i += 32) {
        xq[i] = (_Float16)qsrc[i];
        xm[i] = (_Float16)msrc[i];
    }
    __syncthreads();

    v16h aq[4], am[4];
#pragma unroll
    for (int kk = 0; kk < 4; ++kk) {
        aq[kk] = load_fragA(xq, CC, kk * 32);
        am[kk] = load_fragA(xm, CC, kk * 32);
    }

    const int half = lane >> 4;
    const int nidx = lane & 15;

    for (int h = 0; h < HH; ++h) {
#pragma unroll
        for (int ct = 0; ct < DD / 16; ++ct) {
            v8f accq = {}, acck = {}, accv = {}, accg = {};
            const int wrow = (h * DD + ct * 16) * CC;
#pragma unroll
            for (int kk = 0; kk < 4; ++kk) {
                accq = WMMA_F16(aq[kk], load_fragB(wqt + wrow, CC, kk * 32), accq);
                acck = WMMA_F16(am[kk], load_fragB(wkt + wrow, CC, kk * 32), acck);
                accv = WMMA_F16(am[kk], load_fragB(wvt + wrow, CC, kk * 32), accv);
                accg = WMMA_F16(aq[kk], load_fragB(wgt + wrow, CC, kk * 32), accg);
            }
#pragma unroll
            for (int r = 0; r < 8; ++r) {
                const int m   = half * 8 + r;
                const size_t row = r0 + m;
                const int b   = (int)(row >> 8);          // / N
                const int n   = (int)(row & 255);
                const int d   = ct * 16 + nidx;
                const size_t o = (((size_t)(b * HH + h) * NN) + n) * DD + d;
                qb[o]    = (_Float16)accq[r];
                kb[o]    = (_Float16)acck[r];
                const float g = accg[r] + gbias[h * DD + d];
                gateb[o] = (_Float16)(1.0f / (1.0f + __expf(-g)));
                vtb[(((size_t)(b * HH + h) * DD) + d) * NN + n] = (_Float16)accv[r];
            }
        }
    }
}

// ---------------------------------------------------------------------------
// Kernel 2: attention. 4 waves (64 q-rows) per workgroup, one (b,h) slab of
// K (256x128) and V^T (128x256) staged in LDS ONCE via the Tensor Data Mover
// (CDNA5 TDM, TENSORcnt) and shared by all waves. Per-wave 16KB f32 logits
// scoreboard is reused in place as the f16 P matrix.
// ---------------------------------------------------------------------------
__global__ __launch_bounds__(128) void attn_kernel(
    const _Float16* __restrict__ qb, const _Float16* __restrict__ kb,
    const _Float16* __restrict__ vtb, const _Float16* __restrict__ gateb,
    const float* __restrict__ bias, const float* __restrict__ nb,
    _Float16* __restrict__ wab) {
    __shared__ _Float16 s_k[NN * DD];     // 64 KB: K slab  (256 rows x 128, c contig)
    __shared__ _Float16 s_vt[DD * NN];    // 64 KB: V^T slab (128 rows x 256, k contig)
    __shared__ float    s_s[4][16 * NN];  // 64 KB: per-wave logits (aliased as f16 P)
    __shared__ float    s_rinv[4][16];

    const int bid  = blockIdx.x;          // B*H*(N/64) = 4096
    const int qblk = bid & 3;
    const int h    = (bid >> 2) & 3;
    const int b    = bid >> 4;
    const int wave = threadIdx.x >> 5;
    const int lane = threadIdx.x & 31;
    const int half = lane >> 4;
    const int nidx = lane & 15;
    const int qt   = qblk * 4 + wave;     // q tile (16 rows) index, 0..15

    const size_t bh = (size_t)(b * HH + h);

    // --- TDM: one wave DMAs the K and V^T slabs for this (b,h) into LDS ---
    if (wave == 0) {
        tdm_load_2d_f16((unsigned)(size_t)(void*)s_k,  kb  + bh * NN * DD, DD, NN, DD);
        tdm_load_2d_f16((unsigned)(size_t)(void*)s_vt, vtb + bh * DD * NN, NN, DD, NN);
        __builtin_amdgcn_s_wait_tensorcnt(0);
    }
    __syncthreads();

    // q fragments: 16 x 128, K = channel dim
    const _Float16* qtile = qb + (bh * NN + qt * 16) * DD;
    v16h qf[4];
#pragma unroll
    for (int kk = 0; kk < 4; ++kk) qf[kk] = load_fragA(qtile, DD, kk * 32);

    float* srow = &s_s[wave][0];

    // ---- logits pass: 16 k-tiles of 16, K slab read from LDS ----
    for (int kt = 0; kt < NN / 16; ++kt) {
        v8f acc = {};
        const _Float16* ktile = s_k + (size_t)kt * 16 * DD;
#pragma unroll
        for (int kk = 0; kk < 4; ++kk)
            acc = WMMA_F16(qf[kk], load_fragB(ktile, DD, kk * 32), acc);
        // prefetch next tile's bias rows (streams once from HBM)
        if (kt + 1 < NN / 16) {
            const int qi0 = qt * 16 + half * 8;
            __builtin_prefetch(&bias[((size_t)b * NN + qi0) * NN + (kt + 1) * 16], 0, 0);
        }
#pragma unroll
        for (int r = 0; r < 8; ++r) {
            const int m  = half * 8 + r;
            const int qi = qt * 16 + m;
            const int ki = kt * 16 + nidx;
            float v = acc[r];
            v += bias[((size_t)b * NN + qi) * NN + ki];
            v += nb[((size_t)h * NN + qi) * NN + ki];
            srow[m * NN + ki] = v;
        }
    }

    // ---- softmax: wave32 cooperates on each of its 16 rows; write f16 P
    //      back into the same buffer (row r of P only touches logit rows <= r,
    //      already consumed by the lockstep wave) ----
    _Float16* prow = (_Float16*)srow;
    for (int r = 0; r < 16; ++r) {
        float vals[NN / 32];
        float mx = -3.0e38f;
#pragma unroll
        for (int j = 0; j < NN / 32; ++j) {
            vals[j] = srow[r * NN + lane + 32 * j];
            mx = fmaxf(mx, vals[j]);
        }
#pragma unroll
        for (int i = 16; i >= 1; i >>= 1) mx = fmaxf(mx, __shfl_xor(mx, i, 32));
        float sum = 0.0f;
#pragma unroll
        for (int j = 0; j < NN / 32; ++j) {
            float e = __expf(vals[j] - mx);
            sum += e;
            prow[r * NN + lane + 32 * j] = (_Float16)e;   // row-scale deferred
        }
#pragma unroll
        for (int i = 16; i >= 1; i >>= 1) sum += __shfl_xor(sum, i, 32);
        if (lane == 0) s_rinv[wave][r] = 1.0f / sum;
    }

    // ---- P*V: K = 256 over keys, V^T slab read from LDS ----
    v16h pf[8];
#pragma unroll
    for (int kk = 0; kk < 8; ++kk) pf[kk] = load_fragA(prow, NN, kk * 32);

#pragma unroll
    for (int ct = 0; ct < DD / 16; ++ct) {
        v8f acc = {};
        const _Float16* vtile = s_vt + (size_t)ct * 16 * NN;
#pragma unroll
        for (int kk = 0; kk < 8; ++kk)
            acc = WMMA_F16(pf[kk], load_fragB(vtile, NN, kk * 32), acc);
#pragma unroll
        for (int r = 0; r < 8; ++r) {
            const int m  = half * 8 + r;
            const int qi = qt * 16 + m;
            const int d  = ct * 16 + nidx;
            const float g = (float)gateb[(bh * NN + qi) * DD + d];
            const float val = acc[r] * s_rinv[wave][m] * g;
            wab[((size_t)(b * NN + qi)) * (HH * DD) + h * DD + d] = (_Float16)val;
        }
    }
}

// ---------------------------------------------------------------------------
// Kernel 3: output projection. (B*N, 512) x (512, 128) + bias -> f32 out.
// ---------------------------------------------------------------------------
__global__ __launch_bounds__(32) void outproj_kernel(
    const _Float16* __restrict__ wab, const _Float16* __restrict__ w2t,
    const float* __restrict__ ob, float* __restrict__ out) {
    const size_t r0 = (size_t)blockIdx.x * 16;
    const int lane = threadIdx.x;
    const int half = lane >> 4;
    const int nidx = lane & 15;
    const int K = HH * DD;                                // 512

    v8f acc[OO / 16];
#pragma unroll
    for (int ct = 0; ct < OO / 16; ++ct) acc[ct] = (v8f){};

    const _Float16* atile = wab + r0 * K;
    for (int kk = 0; kk < K / 32; ++kk) {
        v16h a = load_fragA(atile, K, kk * 32);
#pragma unroll
        for (int ct = 0; ct < OO / 16; ++ct)
            acc[ct] = WMMA_F16(a, load_fragB(w2t + ct * 16 * K, K, kk * 32), acc[ct]);
    }
#pragma unroll
    for (int ct = 0; ct < OO / 16; ++ct) {
#pragma unroll
        for (int r = 0; r < 8; ++r) {
            const int m = half * 8 + r;
            const int o = ct * 16 + nidx;
            out[(r0 + m) * OO + o] = acc[ct][r] + ob[o];
        }
    }
}

// ---------------------------------------------------------------------------
extern "C" void kernel_launch(void* const* d_in, const int* in_sizes, int n_in,
                              void* d_out, int out_size, void* d_ws, size_t ws_size,
                              hipStream_t stream) {
    (void)in_sizes; (void)n_in; (void)out_size; (void)ws_size;
    const float* qd = (const float*)d_in[0];   // q_data   (B,N,C)
    const float* md = (const float*)d_in[1];   // m_data   (B,N,C)
    const float* bs = (const float*)d_in[2];   // bias     (B,1,N,N)
    const float* nb = (const float*)d_in[3];   // nonbatched_bias (H,N,N)
    const float* qw = (const float*)d_in[4];   // query_w  (C,H,D)
    const float* kw = (const float*)d_in[5];   // key_w
    const float* vw = (const float*)d_in[6];   // value_w
    const float* gw = (const float*)d_in[7];   // gating_w
    const float* gb = (const float*)d_in[8];   // gating_b (H,D)
    const float* ow = (const float*)d_in[9];   // output_w (H,D,O)
    const float* ob = (const float*)d_in[10];  // output_b (O)
    float* out = (float*)d_out;

    // workspace carve-out (256B aligned)
    char* ws = (char*)d_ws;
    auto alloc = [&](size_t bytes) -> char* {
        char* p = ws;
        ws += (bytes + 255) & ~(size_t)255;
        return p;
    };
    const size_t WSZ   = (size_t)CC * HH * DD;           // 65536
    const size_t QKV   = (size_t)BB * HH * NN * DD;      // 33.5M elems
    const size_t WASZ  = (size_t)BB * NN * HH * DD;

    _Float16* wqt   = (_Float16*)alloc(WSZ * 2);
    _Float16* wkt   = (_Float16*)alloc(WSZ * 2);
    _Float16* wvt   = (_Float16*)alloc(WSZ * 2);
    _Float16* wgt   = (_Float16*)alloc(WSZ * 2);
    _Float16* w2t   = (_Float16*)alloc(WSZ * 2);
    _Float16* qb    = (_Float16*)alloc(QKV * 2);
    _Float16* kb    = (_Float16*)alloc(QKV * 2);
    _Float16* vtb   = (_Float16*)alloc(QKV * 2);
    _Float16* gateb = (_Float16*)alloc(QKV * 2);
    _Float16* wab   = (_Float16*)alloc(WASZ * 2);

    prep_weights<<<WSZ / 256, 256, 0, stream>>>(qw, kw, vw, gw, ow,
                                                wqt, wkt, wvt, wgt, w2t);
    proj_kernel<<<(BB * NN) / 16, 32, 0, stream>>>(qd, md, wqt, wkt, wvt, wgt, gb,
                                                   qb, kb, vtb, gateb);
    attn_kernel<<<BB * HH * (NN / 64), 128, 0, stream>>>(qb, kb, vtb, gateb, bs, nb, wab);
    outproj_kernel<<<(BB * NN) / 16, 32, 0, stream>>>(wab, w2t, ob, out);
}
